// YoutuAttention_38852274160233
// MI455X (gfx1250) — matrix operand order repaired and checked by reading, not compiled, not run on hardware
//
#include <hip/hip_runtime.h>
#include <hip/hip_bf16.h>

typedef __attribute__((ext_vector_type(16))) __bf16 v16bf;
typedef __attribute__((ext_vector_type(8)))  float  v8f;
typedef __attribute__((ext_vector_type(8)))  unsigned short u16x8;

union FragU { u16x8 h[2]; v16bf v; };

__device__ __forceinline__ unsigned short f2bf(float f) {
  unsigned int u = __float_as_uint(f);
  u += 0x7FFFu + ((u >> 16) & 1u);          // round-to-nearest-even
  return (unsigned short)(u >> 16);
}

// ---------------------------------------------------------------------------
// Generic batched GEMM:  C(M,N) = A(M,K) * B   (bf16 WMMA, f32 accum)
//   a(m,k) = A[bz*batchA + m*lda + k]            (f32, or bf16 when A_BF16)
//   b(k,n) = B[bz*batchB + k*ldb_k + n*ldb_n]    (f32 -> bf16)
//   c(m,n) = C[bz*batchC + m*ldc + n]            (f32, or bf16 when C_BF16)
// Block: 256 threads = 8 waves (2x4); block tile 128x128; K-step 32.
// bf16-A tiles are staged with global_load_async_to_lds_b128 (ASYNCcnt).
// GUARD_N only for N not a multiple of 128.
// ---------------------------------------------------------------------------
template <int A_BF16, int C_BF16, int GUARD_N>
__global__ __launch_bounds__(256)
void gemm_wmma_t(const void* __restrict__ Ap, long lda, long batchA,
                 const float* __restrict__ B, long ldb_k, long ldb_n, long batchB,
                 void* __restrict__ Cp, long ldc, long batchC,
                 int M, int N, int K)
{
  __shared__ __attribute__((aligned(16))) unsigned short As[128][40];
  __shared__ __attribute__((aligned(16))) unsigned short Bs[128][40];

  const int tid  = threadIdx.x;
  const int bm   = blockIdx.y, bn = blockIdx.x, bz = blockIdx.z;
  const int lane = tid & 31, w = tid >> 5;
  const int wm   = w >> 2, wn = w & 3;         // 2 x 4 wave layout
  const int lr   = lane & 15, hf = lane >> 4;  // wave32 halves
  const int kbA  = hf ? 8 : 0;                 // A frag: lanes16-31 hold K=8..15 / 24..31
  const int kbB  = hf ? 8 : 0;                 // B frag: lanes16-31 hold K=8..15 / 24..31

  const float* Af = (const float*)Ap;
  const unsigned short* Ah = (const unsigned short*)Ap;
  const long aoff = (long)bz * batchA;
  const long boff = (long)bz * batchB;
  const long coff = (long)bz * batchC;

  v8f acc[4][2];
#pragma unroll
  for (int i = 0; i < 4; i++)
#pragma unroll
    for (int j = 0; j < 2; j++)
#pragma unroll
      for (int e = 0; e < 8; e++) acc[i][j][e] = 0.f;

  for (int k0 = 0; k0 < K; k0 += 32) {
    __syncthreads();
    // ---- stage A tile (128 x 32) ----
    if constexpr (A_BF16) {
      // direct bf16 source: async DMA global -> LDS, 16B per lane per issue
#pragma unroll
      for (int c0 = 0; c0 < 512; c0 += 256) {
        int c = c0 + tid;
        int m = c >> 2, ko = (c & 3) * 8;
        const unsigned short* g = Ah + aoff + ((long)bm * 128 + m) * lda + k0 + ko;
        unsigned lds_addr = (unsigned)(uintptr_t)&As[m][ko];
        asm volatile("global_load_async_to_lds_b128 %0, %1, off"
                     :: "v"(lds_addr), "v"((unsigned long long)(uintptr_t)g)
                     : "memory");
      }
    } else {
#pragma unroll
      for (int i0 = 0; i0 < 128 * 32; i0 += 256) {
        int idx = i0 + tid;
        int m = idx >> 5, k = idx & 31;
        As[m][k] = f2bf(Af[aoff + ((long)bm * 128 + m) * lda + (k0 + k)]);
      }
      if (k0 + 32 < K)
        __builtin_prefetch(&Af[aoff + ((long)bm * 128 + (tid >> 1)) * lda + k0 + 32], 0, 1);
    }
    // ---- stage B tile as [n][k] (transposed in LDS) ----
#pragma unroll
    for (int i0 = 0; i0 < 128 * 32; i0 += 256) {
      int idx = i0 + tid;
      int n = idx >> 5, k = idx & 31;
      long gn = (long)bn * 128 + n;
      unsigned short hv;
      if constexpr (GUARD_N) {
        hv = 0;
        if (gn < N) hv = f2bf(B[boff + (long)(k0 + k) * ldb_k + gn * ldb_n]);
      } else {
        hv = f2bf(B[boff + (long)(k0 + k) * ldb_k + gn * ldb_n]);
      }
      Bs[n][k] = hv;
    }
    if constexpr (A_BF16)
      asm volatile("s_wait_asynccnt 0x0" ::: "memory");
    __syncthreads();

    FragU afr[4], bfr[2];
#pragma unroll
    for (int mt = 0; mt < 4; mt++) {
      const unsigned short* p = &As[wm * 64 + mt * 16 + lr][0];
      afr[mt].h[0] = *(const u16x8*)(p + kbA);
      afr[mt].h[1] = *(const u16x8*)(p + 16 + kbA);
    }
#pragma unroll
    for (int nt = 0; nt < 2; nt++) {
      const unsigned short* p = &Bs[wn * 32 + nt * 16 + lr][0];
      bfr[nt].h[0] = *(const u16x8*)(p + kbB);
      bfr[nt].h[1] = *(const u16x8*)(p + kbB + 16);
    }
#pragma unroll
    for (int mt = 0; mt < 4; mt++)
#pragma unroll
      for (int nt = 0; nt < 2; nt++)
        acc[mt][nt] = __builtin_amdgcn_wmma_f32_16x16x32_bf16(
            false, afr[mt].v, false, bfr[nt].v, (short)0, acc[mt][nt], false, false);
  }

  float* Cf = (float*)Cp;
  unsigned short* Ch = (unsigned short*)Cp;
#pragma unroll
  for (int mt = 0; mt < 4; mt++)
#pragma unroll
    for (int nt = 0; nt < 2; nt++)
#pragma unroll
      for (int v = 0; v < 8; v++) {
        long m = (long)bm * 128 + wm * 64 + mt * 16 + v + hf * 8;
        long n = (long)bn * 128 + wn * 32 + nt * 16 + lr;
        bool ok = true;
        if constexpr (GUARD_N) ok = (n < N);
        if (ok) {
          long ci = coff + m * ldc + n;
          if constexpr (C_BF16) Ch[ci] = f2bf(acc[mt][nt][v]);
          else                  Cf[ci] = acc[mt][nt][v];
        }
      }
}

// ---------------------------------------------------------------------------
// RMS-norm of kv_c, RoPE on q_rope / k_rope, assemble bf16 k_states,
// transposed kT[d][t] (V path), and rope part of q_states. One block per token.
// ---------------------------------------------------------------------------
__global__ __launch_bounds__(256)
void rope_rms_kernel(const int* __restrict__ positions,
                     const float* __restrict__ kvlat,   // T x 576 f32
                     const float* __restrict__ lnw,     // 512
                     const float* __restrict__ qraw,    // T x 3072 f32
                     unsigned short* __restrict__ kst,  // T x 576 bf16
                     unsigned short* __restrict__ kT,   // 512 x T bf16
                     unsigned short* __restrict__ qst,  // T x 16 x 576 bf16
                     int T)
{
  const int t = blockIdx.x, tid = threadIdx.x;
  __shared__ float red[256];
  __shared__ float cs[32], sn[32];
  const float* kv = kvlat + (long)t * 576;

  float ss = 0.f;
#pragma unroll
  for (int j0 = 0; j0 < 512; j0 += 256) { float v = kv[j0 + tid]; ss += v * v; }
  red[tid] = ss;
  __syncthreads();
  for (int s = 128; s > 0; s >>= 1) {
    if (tid < s) red[tid] += red[tid + s];
    __syncthreads();
  }
  const float rs = rsqrtf(red[0] * (1.f / 512.f) + 1e-6f);

  const float pos = (float)positions[t];
  if (tid < 32) {
    float inv = __powf(10000.f, -(2.f * tid) / 64.f);
    float f = pos * inv;
    cs[tid] = cosf(f);
    sn[tid] = sinf(f);
  }
  __syncthreads();

#pragma unroll
  for (int j0 = 0; j0 < 512; j0 += 256) {
    int j = j0 + tid;
    unsigned short hv = f2bf(kv[j] * rs * lnw[j]);
    kst[(long)t * 576 + j] = hv;
    kT[(long)j * T + t]    = hv;    // V is kv_c: keep key-contiguous copy
  }
  if (tid < 32) {
    float x1 = kv[512 + tid], x2 = kv[544 + tid];
    kst[(long)t * 576 + 512 + tid] = f2bf(x1 * cs[tid] - x2 * sn[tid]);
    kst[(long)t * 576 + 544 + tid] = f2bf(x2 * cs[tid] + x1 * sn[tid]);
  }
#pragma unroll
  for (int w0 = 0; w0 < 512; w0 += 256) {
    int widx = w0 + tid;
    int h = widx >> 5, i = widx & 31;
    const float* qb = qraw + (long)t * 3072 + h * 192 + 128;
    float x1 = qb[i], x2 = qb[32 + i];
    unsigned short* q = qst + (long)t * 9216 + h * 576 + 512;
    q[i]      = f2bf(x1 * cs[i] - x2 * sn[i]);
    q[32 + i] = f2bf(x2 * cs[i] + x1 * sn[i]);
  }
}

// ---------------------------------------------------------------------------
// Causal flash attention. Grid (T/16, H), 128 threads = 4 waves.
// All 4 waves cooperate on S = Q K^T: every wave statically takes 4 of the
// 18 d-steps (ds = w*4 + i), waves 0/1 take the 2 leftovers (ds 16/17);
// partial S is reduced through LDS. Each wave then owns one 128-dim quarter
// of the 512-dim V space for PV. K/V fragments stream from global: k_states
// is 2.4 MB -> resident in the 192 MB L2, so LDS is spent only on the
// S-reduce and P-transpose exchanges. All short loops force-unrolled so
// vector-element accesses stay in static registers (no v_movrels).
// ---------------------------------------------------------------------------
__global__ __launch_bounds__(128)
void mla_flash_kernel(const unsigned short* __restrict__ qst, // T x 16 x 576
                      const unsigned short* __restrict__ kst, // T x 576
                      const unsigned short* __restrict__ kT,  // 512 x T
                      unsigned short* __restrict__ attn_o,    // T x 16 x 512
                      int T)
{
  __shared__ __attribute__((aligned(16))) float Sred[4][2][32][8];       // 8 KB
  __shared__ __attribute__((aligned(16))) unsigned short Ps[4][16][40];  // 5 KB

  const int tid  = threadIdx.x;
  const int w    = tid >> 5, lane = tid & 31;
  const int lr   = lane & 15, hf = lane >> 4;
  const int kbA  = hf ? 8 : 0;
  const int kbB  = hf ? 8 : 0;
  const int rowbase = blockIdx.x * 16;
  const int h    = blockIdx.y;
  const int vq   = w;                         // V quarter: [vq*128, vq*128+128)

  const unsigned short* qrow = qst + ((long)(rowbase + lr) * 16 + h) * 576;
  const float scale = 0.07216878364870323f;   // (128+64)^-0.5

  v8f acc[8];
#pragma unroll
  for (int nt = 0; nt < 8; nt++)
#pragma unroll
    for (int e = 0; e < 8; e++) acc[nt][e] = 0.f;
  float rmax[8], rsum[8];
#pragma unroll
  for (int v = 0; v < 8; v++) { rmax[v] = -3.0e38f; rsum[v] = 0.f; }

  const int kmax = rowbase + 15;              // block-uniform trip count
  for (int kt = 0; kt <= kmax; kt += 32) {
    // ---- partial S = Q K^T over keys [kt, kt+32), this wave's d-slice ----
    v8f sac[2];
#pragma unroll
    for (int nb = 0; nb < 2; nb++)
#pragma unroll
      for (int e = 0; e < 8; e++) sac[nb][e] = 0.f;
    const unsigned short* krow0 = kst + (long)(kt + lr) * 576;
    const unsigned short* krow1 = kst + (long)(kt + 16 + lr) * 576;
#pragma unroll
    for (int i = 0; i < 4; i++) {
      int d = (w * 4 + i) * 32;
      FragU a, b0, b1;
      a.h[0]  = *(const u16x8*)(qrow + d + kbA);
      a.h[1]  = *(const u16x8*)(qrow + d + 16 + kbA);
      b0.h[0] = *(const u16x8*)(krow0 + d + kbB);
      b0.h[1] = *(const u16x8*)(krow0 + d + kbB + 16);
      b1.h[0] = *(const u16x8*)(krow1 + d + kbB);
      b1.h[1] = *(const u16x8*)(krow1 + d + kbB + 16);
      sac[0] = __builtin_amdgcn_wmma_f32_16x16x32_bf16(false, a.v, false, b0.v, (short)0, sac[0], false, false);
      sac[1] = __builtin_amdgcn_wmma_f32_16x16x32_bf16(false, a.v, false, b1.v, (short)0, sac[1], false, false);
    }
    if (w < 2) {                 // leftover d-steps 16, 17 (static per wave)
      int d = (16 + w) * 32;
      FragU a, b0, b1;
      a.h[0]  = *(const u16x8*)(qrow + d + kbA);
      a.h[1]  = *(const u16x8*)(qrow + d + 16 + kbA);
      b0.h[0] = *(const u16x8*)(krow0 + d + kbB);
      b0.h[1] = *(const u16x8*)(krow0 + d + kbB + 16);
      b1.h[0] = *(const u16x8*)(krow1 + d + kbB);
      b1.h[1] = *(const u16x8*)(krow1 + d + kbB + 16);
      sac[0] = __builtin_amdgcn_wmma_f32_16x16x32_bf16(false, a.v, false, b0.v, (short)0, sac[0], false, false);
      sac[1] = __builtin_amdgcn_wmma_f32_16x16x32_bf16(false, a.v, false, b1.v, (short)0, sac[1], false, false);
    }
    if (kt + 32 <= kmax)
      __builtin_prefetch(kst + (long)(kt + 32 + lr) * 576, 0, 1);

    // ---- cross-wave S reduction through LDS ----
    *(v8f*)&Sred[w][0][lane][0] = sac[0];
    *(v8f*)&Sred[w][1][lane][0] = sac[1];
    __syncthreads();
    v8f s0 = *(const v8f*)&Sred[0][0][lane][0];
    v8f s1 = *(const v8f*)&Sred[0][1][lane][0];
#pragma unroll
    for (int wp = 1; wp < 4; wp++) {
      s0 += *(const v8f*)&Sred[wp][0][lane][0];
      s1 += *(const v8f*)&Sred[wp][1][lane][0];
    }
    __syncthreads();   // Sred reusable next iteration

    // ---- causal mask + online softmax (rows live per-lane in C layout) ----
#pragma unroll
    for (int v = 0; v < 8; v++) {
      int row = rowbase + v + hf * 8;
      float e0 = (kt + lr      <= row) ? s0[v] * scale : -3.0e38f;
      float e1 = (kt + 16 + lr <= row) ? s1[v] * scale : -3.0e38f;
      float tm = fmaxf(e0, e1);
#pragma unroll
      for (int m = 1; m < 16; m <<= 1) tm = fmaxf(tm, __shfl_xor(tm, m, 32));
      float nm = fmaxf(rmax[v], tm);
      float so = __expf(rmax[v] - nm);
      float p0 = __expf(e0 - nm);
      float p1 = __expf(e1 - nm);
      float ts = p0 + p1;
#pragma unroll
      for (int m = 1; m < 16; m <<= 1) ts += __shfl_xor(ts, m, 32);
      rsum[v] = rsum[v] * so + ts;
      rmax[v] = nm;
#pragma unroll
      for (int nt = 0; nt < 8; nt++) acc[nt][v] *= so;
      Ps[w][v + hf * 8][lr]      = f2bf(p0);
      Ps[w][v + hf * 8][16 + lr] = f2bf(p1);
    }
    asm volatile("s_wait_dscnt 0x0" ::: "memory");  // wave-local LDS fence

    // ---- O += P * V  (V = kv_c rows of kT, key-contiguous) ----
    FragU pa;
    pa.h[0] = *(const u16x8*)(&Ps[w][lr][0] + kbA);
    pa.h[1] = *(const u16x8*)(&Ps[w][lr][0] + 16 + kbA);
#pragma unroll
    for (int nt = 0; nt < 8; nt++) {
      int vd = vq * 128 + nt * 16 + lr;
      const unsigned short* vp = kT + (long)vd * T + kt;
      FragU b;
      b.h[0] = *(const u16x8*)(vp + kbB);
      b.h[1] = *(const u16x8*)(vp + kbB + 16);
      acc[nt] = __builtin_amdgcn_wmma_f32_16x16x32_bf16(false, pa.v, false, b.v, (short)0, acc[nt], false, false);
    }
    asm volatile("s_wait_dscnt 0x0" ::: "memory");  // Ps reused next iter
  }

#pragma unroll
  for (int nt = 0; nt < 8; nt++)
#pragma unroll
    for (int v = 0; v < 8; v++) {
      long m = rowbase + v + hf * 8;
      long r = vq * 128 + nt * 16 + lr;
      attn_o[m * (16L * 512) + (long)h * 512 + r] = f2bf(acc[nt][v] / rsum[v]);
    }
}

// ---------------------------------------------------------------------------
extern "C" void kernel_launch(void* const* d_in, const int* in_sizes, int n_in,
                              void* d_out, int out_size, void* d_ws, size_t ws_size,
                              hipStream_t stream)
{
  const int*   positions = (const int*)  d_in[0];
  const float* hidden    = (const float*)d_in[1];   // T x 2048
  const float* q_proj_w  = (const float*)d_in[2];   // 3072 x 2048
  const float* kv_a_w    = (const float*)d_in[3];   // 576 x 2048
  const float* ln_w      = (const float*)d_in[4];   // 512
  const float* kv_b_w    = (const float*)d_in[5];   // 4096 x 512
  const float* o_proj_w  = (const float*)d_in[6];   // 2048 x 2048
  float* out = (float*)d_out;                        // T x 2048 f32
  const int T = in_sizes[0];

  // workspace layout (with late-stage aliasing over dead buffers)
  char* ws = (char*)d_ws;
  unsigned short* qst = (unsigned short*)ws;                 // T*16*576 bf16
  size_t o1 = (size_t)T * 16 * 576 * 2;
  unsigned short* kst = (unsigned short*)(ws + o1);          // T*576 bf16
  size_t o2 = o1 + (size_t)T * 576 * 2;
  unsigned short* kT  = (unsigned short*)(ws + o2);          // 512*T bf16
  size_t o3 = o2 + (size_t)512 * T * 2;
  float* qraw  = (float*)(ws + o3);                          // T*3072 f32
  size_t o4 = o3 + (size_t)T * 3072 * 4;
  float* kvlat = (float*)(ws + o4);                          // T*576 f32
  unsigned short* attn_o = (unsigned short*)(ws + o3);       // aliases qraw/kvlat (dead)
  float* ov = (float*)ws;                                    // aliases qst (dead)

  dim3 blk(256);

  // 1a: q_raw = hidden @ q_proj_w^T    (M=T, N=3072, K=2048)
  gemm_wmma_t<0, 0, 0><<<dim3(3072 / 128, T / 128, 1), blk, 0, stream>>>(
      hidden, 2048, 0, q_proj_w, 1, 2048, 0, qraw, 3072, 0, T, 3072, 2048);

  // 1b: kv_latent = hidden @ kv_a_w^T  (M=T, N=576, K=2048; N guarded)
  gemm_wmma_t<0, 0, 1><<<dim3((576 + 127) / 128, T / 128, 1), blk, 0, stream>>>(
      hidden, 2048, 0, kv_a_w, 1, 2048, 0, kvlat, 576, 0, T, 576, 2048);

  // 2: rms-norm + rope + assemble k_states / kT / q_states rope part
  rope_rms_kernel<<<dim3(T), blk, 0, stream>>>(positions, kvlat, ln_w, qraw,
                                               kst, kT, qst, T);

  // 3: q_abs[h] = q_nope[h] @ w_key[h]  (batched over 16 heads; M=T,N=512,K=128)
  gemm_wmma_t<0, 1, 0><<<dim3(512 / 128, T / 128, 16), blk, 0, stream>>>(
      qraw, 3072, 192,
      kv_b_w, 512, 1, 256L * 512,
      qst, 16L * 576, 576,
      T, 512, 128);

  // 4: causal flash attention -> attn_o bf16 (T,16,512)
  mla_flash_kernel<<<dim3(T / 16, 16), dim3(128), 0, stream>>>(qst, kst, kT, attn_o, T);

  // 5: o_v[h] = attn_o[h] @ w_vo[h]^T  (batched; M=T, N=128, K=512; async-A)
  gemm_wmma_t<1, 0, 0><<<dim3(1, T / 128, 16), blk, 0, stream>>>(
      attn_o, 16L * 512, 512,
      kv_b_w + 128L * 512, 1, 512, 256L * 512,
      ov, 2048, 128,
      T, 128, 512);

  // 6: out = ov @ o_proj_w^T  (M=T, N=2048, K=2048)
  gemm_wmma_t<0, 0, 0><<<dim3(2048 / 128, T / 128, 1), blk, 0, stream>>>(
      ov, 2048, 0, o_proj_w, 1, 2048, 0, out, 2048, 0, T, 2048, 2048);
}
